// FastAttention_32418413150347
// MI455X (gfx1250) — matrix operand order repaired and checked by reading, not compiled, not run on hardware
//
#include <hip/hip_runtime.h>
#include <math.h>

typedef __attribute__((ext_vector_type(16))) __bf16 v16bf;
typedef __attribute__((ext_vector_type(8)))  __bf16 v8bf;
typedef __attribute__((ext_vector_type(8)))  float  v8f;

#define N_CTX 4096
#define D_DIM 64
#define M_DIM 256
#define NBH   64
#define DATA_NORM 0.35355339059327373f   /* 64^-0.25 */
#define RATIO     0.0625f                /* 256^-0.5 */
#define KEPS      1e-4f

// ---------------- WMMA fragment helpers (wave32, 16x16x32 bf16) ----------------
// A fragment (16x32): lane l<16 -> row l, K k0+0..15;
// lane l+16 -> row l, K k0+16..31. Two contiguous 16B reads.
__device__ __forceinline__ v16bf load_a_frag(const __bf16* base, int stride, int k0, int lane) {
  int r = lane & 15, hi = lane >> 4;
  const __bf16* p = base + r * stride + k0 + hi * 16;
  union { v16bf v; v8bf h[2]; } u;
  u.h[0] = *(const v8bf*)(p);
  u.h[1] = *(const v8bf*)(p + 8);
  return u.v;
}

// B fragment (32x16) from a transposed (N-major) operand: B[k][n] = baseT[(n0+n)*stride + k].
// lane n<16 -> col n, K k0+0..15 ; lane n+16 -> col n, K k0+16..31. One 32B read.
__device__ __forceinline__ v16bf load_bT_frag(const __bf16* baseT, int stride, int n0, int k0, int lane) {
  int n = lane & 15, hi = lane >> 4;
  return *(const v16bf*)(baseT + (n0 + n) * stride + k0 + hi * 16);
}

__device__ __forceinline__ v8f wmma_bf16(v16bf a, v16bf b, v8f c) {
  return __builtin_amdgcn_wmma_f32_16x16x32_bf16(false, a, false, b, (short)0, c, false, false);
}

// ---------------- kernel 0: zero scratch, convert P -> bf16 ----------------
__global__ __launch_bounds__(256) void prep_kernel(const float* __restrict__ Pm,
                                                   float* __restrict__ ctx,
                                                   float* __restrict__ kcum,
                                                   unsigned short* __restrict__ Pb_u) {
  __bf16* Pb = (__bf16*)Pb_u;
  int stride = gridDim.x * blockDim.x;
  int tid = blockIdx.x * blockDim.x + threadIdx.x;
  for (int i = tid; i < NBH * M_DIM * D_DIM; i += stride) ctx[i] = 0.0f;
  for (int i = tid; i < NBH * M_DIM; i += stride) kcum[i] = 0.0f;
  for (int i = tid; i < M_DIM * D_DIM; i += stride) Pb[i] = (__bf16)Pm[i];
}

// ---------------- kernel 1: partial max of k-dash ----------------
__global__ __launch_bounds__(128) void kmax_partial_kernel(const float* __restrict__ K,
                                                           const unsigned short* __restrict__ Pb_u,
                                                           float* __restrict__ partial) {
  const __bf16* Pb = (const __bf16*)Pb_u;
  __shared__ __bf16 kst[64 * 64];
  __shared__ float wmax[4];
  int t = threadIdx.x, lane = t & 31, w = t >> 5;
  int bh = blockIdx.x >> 6, tile = blockIdx.x & 63;
  const float* kg = K + ((size_t)bh * N_CTX + (size_t)tile * 64) * D_DIM;
  for (int i = t; i < 64 * 64; i += 128) kst[i] = (__bf16)(kg[i] * DATA_NORM);
  __syncthreads();
  const __bf16* myk = kst + w * 16 * 64;
  v16bf a0 = load_a_frag(myk, 64, 0, lane);
  v16bf a1 = load_a_frag(myk, 64, 32, lane);
  float m = -3.0e38f;
  for (int jt = 0; jt < 16; ++jt) {
    v8f c = {};
    c = wmma_bf16(a0, load_bT_frag(Pb, 64, jt * 16, 0, lane), c);
    c = wmma_bf16(a1, load_bT_frag(Pb, 64, jt * 16, 32, lane), c);
#pragma unroll
    for (int r = 0; r < 8; ++r) m = fmaxf(m, c[r]);
  }
#pragma unroll
  for (int off = 16; off > 0; off >>= 1) m = fmaxf(m, __shfl_xor(m, off, 32));
  if (lane == 0) wmax[w] = m;
  __syncthreads();
  if (t == 0) partial[blockIdx.x] = fmaxf(fmaxf(wmax[0], wmax[1]), fmaxf(wmax[2], wmax[3]));
}

// ---------------- kernel 2: reduce partials -> global stab ----------------
__global__ __launch_bounds__(256) void kmax_reduce_kernel(const float* __restrict__ partial, int n,
                                                          float* __restrict__ kmax) {
  __shared__ float sm[8];
  float m = -3.0e38f;
  for (int i = threadIdx.x; i < n; i += 256) m = fmaxf(m, partial[i]);
#pragma unroll
  for (int off = 16; off > 0; off >>= 1) m = fmaxf(m, __shfl_xor(m, off, 32));
  if ((threadIdx.x & 31) == 0) sm[threadIdx.x >> 5] = m;
  __syncthreads();
  if (threadIdx.x == 0) {
    float mm = sm[0];
    for (int i = 1; i < 8; ++i) mm = fmaxf(mm, sm[i]);
    *kmax = mm;
  }
}

// ---------------- kernel 3: k' features + k_cumsum + context = k'^T @ v ----------------
__global__ __launch_bounds__(128) void kctx_kernel(const float* __restrict__ K,
                                                   const float* __restrict__ V,
                                                   const unsigned short* __restrict__ Pb_u,
                                                   const float* __restrict__ kmax_p,
                                                   float* __restrict__ kcum,
                                                   float* __restrict__ ctx) {
  const __bf16* Pb = (const __bf16*)Pb_u;
  __shared__ __bf16 kst[64 * 64];        // k rows (scaled) bf16
  __shared__ __bf16 vT[64 * 64];         // v transposed [e][n_local]
  __shared__ __bf16 kpT[M_DIM * 64];     // k' transposed [m][n_local]
  __shared__ float  diag[64];
  int t = threadIdx.x, lane = t & 31, w = t >> 5;
  int nn = lane & 15, hi = lane >> 4;
  int bh = blockIdx.x >> 3, chunk = blockIdx.x & 7;
  const float* kg = K + ((size_t)bh * N_CTX + (size_t)chunk * 512) * D_DIM;
  const float* vg = V + ((size_t)bh * N_CTX + (size_t)chunk * 512) * D_DIM;
  float kmaxv = *kmax_p;
  float cum0 = 0.f, cum1 = 0.f;
  v8f zero = {};
  v8f acc[4][4];
#pragma unroll
  for (int i = 0; i < 4; ++i)
#pragma unroll
    for (int j = 0; j < 4; ++j) acc[i][j] = zero;

  for (int g = 0; g < 8; ++g) {
    __syncthreads();   // previous group's kpT/vT fully consumed
    for (int i = t; i < 64 * 64; i += 128) {
      kst[i] = (__bf16)(kg[g * 4096 + i] * DATA_NORM);
      int rr = i >> 6, cc = i & 63;
      vT[cc * 64 + rr] = (__bf16)(vg[g * 4096 + i]);
    }
    { // diag for this wave's 16 rows (fp32 from global)
      const float* krow = kg + (size_t)(g * 64 + w * 16 + nn) * 64 + hi * 32;
      float s = 0.f;
#pragma unroll
      for (int i = 0; i < 32; ++i) { float x = krow[i]; s += x * x; }
      s += __shfl_xor(s, 16, 32);
      if (lane < 16) diag[w * 16 + nn] = 0.5f * DATA_NORM * DATA_NORM * s;
    }
    __syncthreads();
    { // dash GEMM + exp -> kpT (transposed store)
      const __bf16* myk = kst + w * 16 * 64;
      v16bf a0 = load_a_frag(myk, 64, 0, lane);
      v16bf a1 = load_a_frag(myk, 64, 32, lane);
      for (int jt = 0; jt < 16; ++jt) {
        v8f c = {};
        c = wmma_bf16(a0, load_bT_frag(Pb, 64, jt * 16, 0, lane), c);
        c = wmma_bf16(a1, load_bT_frag(Pb, 64, jt * 16, 32, lane), c);
#pragma unroll
        for (int r = 0; r < 8; ++r) {
          int row = r + hi * 8;
          float kp = RATIO * (__expf(c[r] - diag[w * 16 + row] - kmaxv) + KEPS);
          kpT[(jt * 16 + nn) * 64 + w * 16 + row] = (__bf16)kp;
        }
      }
    }
    __syncthreads();
    { // k_cumsum accumulation: thread owns 2 m-rows
      const __bf16* r0 = kpT + (t * 2) * 64;
      const __bf16* r1 = r0 + 64;
      float s0 = 0.f, s1 = 0.f;
      for (int n = 0; n < 64; ++n) { s0 += (float)r0[n]; s1 += (float)r1[n]; }
      cum0 += s0; cum1 += s1;
    }
    // context accumulation: wave w owns m-tiles 4w..4w+3, e-tiles 0..3, K = 64 rows
#pragma unroll
    for (int kt = 0; kt < 2; ++kt) {
      v16bf b0 = load_bT_frag(vT, 64, 0,  kt * 32, lane);
      v16bf b1 = load_bT_frag(vT, 64, 16, kt * 32, lane);
      v16bf b2 = load_bT_frag(vT, 64, 32, kt * 32, lane);
      v16bf b3 = load_bT_frag(vT, 64, 48, kt * 32, lane);
#pragma unroll
      for (int mi = 0; mi < 4; ++mi) {
        v16bf a = load_a_frag(kpT + ((w * 4 + mi) * 16) * 64, 64, kt * 32, lane);
        acc[mi][0] = wmma_bf16(a, b0, acc[mi][0]);
        acc[mi][1] = wmma_bf16(a, b1, acc[mi][1]);
        acc[mi][2] = wmma_bf16(a, b2, acc[mi][2]);
        acc[mi][3] = wmma_bf16(a, b3, acc[mi][3]);
      }
    }
  }
  atomicAdd(&kcum[bh * M_DIM + t * 2 + 0], cum0);
  atomicAdd(&kcum[bh * M_DIM + t * 2 + 1], cum1);
  float* ctxb = ctx + (size_t)bh * M_DIM * D_DIM;
#pragma unroll
  for (int mi = 0; mi < 4; ++mi)
#pragma unroll
    for (int et = 0; et < 4; ++et)
#pragma unroll
      for (int r = 0; r < 8; ++r) {
        int mrow = (w * 4 + mi) * 16 + r + hi * 8;
        atomicAdd(&ctxb[mrow * 64 + et * 16 + nn], acc[mi][et][r]);
      }
}

// ---------------- kernel 4: q' features + D_inv + out = (q' @ ctx) * D_inv ----------------
__global__ __launch_bounds__(64) void qout_kernel(const float* __restrict__ Q,
                                                  const unsigned short* __restrict__ Pb_u,
                                                  const float* __restrict__ kcum,
                                                  const float* __restrict__ ctx,
                                                  float* __restrict__ out) {
  const __bf16* Pb = (const __bf16*)Pb_u;
  __shared__ __bf16 ctxT[64 * 256];     // context transposed [e][m], 32KB
  __shared__ __bf16 qst[2][16 * 64];    // per-wave q rows (scaled)
  __shared__ __bf16 qp[2][16 * 256];    // per-wave q' strip
  __shared__ float  diagl[2][16];
  __shared__ float  dinv[2][16];
  int t = threadIdx.x, lane = t & 31, w = t >> 5;
  int nn = lane & 15, hi = lane >> 4;
  int bh = blockIdx.x >> 7, tile = blockIdx.x & 127;
  int n0 = tile * 32;
  const float* qg = Q + ((size_t)bh * N_CTX + n0) * D_DIM;
  const float* ctxg = ctx + (size_t)bh * M_DIM * D_DIM;
  const float* kc = kcum + bh * M_DIM;

  for (int i = t; i < M_DIM * D_DIM; i += 64) {
    int m = i >> 6, e = i & 63;
    ctxT[e * 256 + m] = (__bf16)ctxg[i];
  }
  for (int i = t; i < 32 * 64; i += 64) {
    int rr = i >> 6;
    qst[rr >> 4][(rr & 15) * 64 + (i & 63)] = (__bf16)(qg[i] * DATA_NORM);
  }
  __syncthreads();

  const __bf16* myq = qst[w];
  { // diag (fp32 from global)
    const float* qr = qg + (size_t)(w * 16 + nn) * 64 + hi * 32;
    float s = 0.f;
#pragma unroll
    for (int i = 0; i < 32; ++i) { float x = qr[i]; s += x * x; }
    s += __shfl_xor(s, 16, 32);
    if (lane < 16) diagl[w][nn] = 0.5f * DATA_NORM * DATA_NORM * s;
  }
  v16bf a0 = load_a_frag(myq, 64, 0, lane);
  v16bf a1 = load_a_frag(myq, 64, 32, lane);

  // pass 1: row-wise max of dash (no dash buffer)
  float rm[8];
#pragma unroll
  for (int r = 0; r < 8; ++r) rm[r] = -3.0e38f;
  for (int jt = 0; jt < 16; ++jt) {
    v8f c = {};
    c = wmma_bf16(a0, load_bT_frag(Pb, 64, jt * 16, 0, lane), c);
    c = wmma_bf16(a1, load_bT_frag(Pb, 64, jt * 16, 32, lane), c);
#pragma unroll
    for (int r = 0; r < 8; ++r) {
      float m = c[r];
      m = fmaxf(m, __shfl_xor(m, 1, 32));
      m = fmaxf(m, __shfl_xor(m, 2, 32));
      m = fmaxf(m, __shfl_xor(m, 4, 32));
      m = fmaxf(m, __shfl_xor(m, 8, 32));
      rm[r] = fmaxf(rm[r], m);
    }
  }
  // pass 2: recompute dash, exp -> q' strip in LDS
  for (int jt = 0; jt < 16; ++jt) {
    v8f c = {};
    c = wmma_bf16(a0, load_bT_frag(Pb, 64, jt * 16, 0, lane), c);
    c = wmma_bf16(a1, load_bT_frag(Pb, 64, jt * 16, 32, lane), c);
#pragma unroll
    for (int r = 0; r < 8; ++r) {
      int row = r + hi * 8;
      float v = RATIO * (__expf(c[r] - diagl[w][row] - rm[r]) + KEPS);
      qp[w][row * 256 + jt * 16 + nn] = (__bf16)v;
    }
  }
  __syncthreads();
  { // D_inv = 1 / (q' . k_cumsum)   (fp32 accumulate)
    float s = 0.f;
    const __bf16* qr = qp[w] + nn * 256 + hi * 128;
    const float* kch = kc + hi * 128;
    for (int i = 0; i < 128; ++i) s += (float)qr[i] * kch[i];
    s += __shfl_xor(s, 16, 32);
    if (lane < 16) dinv[w][nn] = 1.0f / s;
  }
  __syncthreads();
  // out GEMM: [16 x 256] x [256 x 64]
  float* og = out + ((size_t)bh * N_CTX + n0 + w * 16) * 64;
#pragma unroll
  for (int et = 0; et < 4; ++et) {
    v8f c = {};
#pragma unroll
    for (int kt = 0; kt < 8; ++kt) {
      v16bf a = load_a_frag(qp[w], 256, kt * 32, lane);
      v16bf b = load_bT_frag(ctxT, 256, et * 16, kt * 32, lane);
      c = wmma_bf16(a, b, c);
    }
#pragma unroll
    for (int r = 0; r < 8; ++r) {
      int row = r + hi * 8;
      og[(size_t)row * 64 + et * 16 + nn] = c[r] * dinv[w][row];
    }
  }
}

// ---------------- host ----------------
extern "C" void kernel_launch(void* const* d_in, const int* in_sizes, int n_in,
                              void* d_out, int out_size, void* d_ws, size_t ws_size,
                              hipStream_t stream) {
  const float* q  = (const float*)d_in[0];
  const float* k  = (const float*)d_in[1];
  const float* v  = (const float*)d_in[2];
  const float* Pm = (const float*)d_in[3];
  float* out = (float*)d_out;

  float* wsf     = (float*)d_ws;
  float* kmax    = wsf;                       // 1 float
  float* partial = wsf + 64;                  // 4096 floats
  float* kcum    = wsf + 64 + 4096;           // 64*256 floats
  float* ctx     = kcum + NBH * M_DIM;        // 64*256*64 floats
  unsigned short* Pb = (unsigned short*)(ctx + NBH * M_DIM * D_DIM);  // 256*64 bf16

  prep_kernel<<<1024, 256, 0, stream>>>(Pm, ctx, kcum, Pb);
  kmax_partial_kernel<<<NBH * (N_CTX / 64), 128, 0, stream>>>(k, Pb, partial);
  kmax_reduce_kernel<<<1, 256, 0, stream>>>(partial, NBH * (N_CTX / 64), kmax);
  kctx_kernel<<<NBH * (N_CTX / 512), 128, 0, stream>>>(k, v, Pb, kmax, kcum, ctx);
  qout_kernel<<<NBH * (N_CTX / 32), 64, 0, stream>>>(q, Pb, kcum, ctx, out);
}
